// MCGNet_87960930222297
// MI455X (gfx1250) — matrix operand order repaired and checked by reading, not compiled, not run on hardware
//
#include <hip/hip_runtime.h>
#include <hip/hip_bf16.h>
#include <stdint.h>

// ---------------- CDNA5 WMMA types ----------------
typedef __attribute__((ext_vector_type(16))) __bf16 v16bf;
typedef __attribute__((ext_vector_type(8)))  float  v8f;

union V16BF { uint4 q[2]; v16bf v; };

__device__ __forceinline__ unsigned short f2bf(float f) {
  union { float f; unsigned u; } c; c.f = f;
  unsigned u = c.u;
  u += 0x7FFFu + ((u >> 16) & 1u);   // round-to-nearest-even to bf16
  return (unsigned short)(u >> 16);
}

__device__ __forceinline__ float leaky(float v) { return v >= 0.f ? v : 0.01f * v; }

#define BN_SCALE rsqrtf(1.f + 1e-5f)

// ---------------- WMMA GEMM: C[N,MT*16] = act(A[N,K]bf16 @ Bt[MT*16,K]bf16^T + bias)
// grid = (N/128), block = 256 (8 waves); wave w -> rows [row0, row0+16).
// Each wave keeps A fragments in registers and sweeps all MT column tiles,
// so the (large) A operand is read exactly once per k-step.
template<int MT>
__global__ __launch_bounds__(256)
void wmma_gemm_mt(const unsigned short* __restrict__ A,
                  const unsigned short* __restrict__ Bt,
                  const float* __restrict__ bias,
                  float* __restrict__ C,
                  int N, int K, int act) {
  const int Mo   = MT * 16;
  const int lane = threadIdx.x & 31;
  const int wave = threadIdx.x >> 5;
  const int half = lane >> 4;
  const int rl   = lane & 15;
  const int row0 = blockIdx.x * 128 + wave * 16;

  const unsigned short* arow = A + (size_t)(row0 + rl) * K;

  v8f acc[MT];
#pragma unroll
  for (int t = 0; t < MT; ++t) { v8f z = {}; acc[t] = z; }

  for (int k = 0; k < K; k += 32) {
    __builtin_prefetch(arow + k + 64, 0, 0);   // global_prefetch_b8
    V16BF a;
    // 16-bit A 16x32 layout: lane half h holds K in [k+8h, k+8h+8) and [k+16+8h, k+16+8h+8)
    a.q[0] = *(const uint4*)(arow + k + half * 8);
    a.q[1] = *(const uint4*)(arow + k + 16 + half * 8);
#pragma unroll
    for (int t = 0; t < MT; ++t) {
      // 16-bit B 32x16 layout: lane half h holds K in [k+8h, k+8h+8) and [k+16+8h, k+16+8h+8)
      const unsigned short* brow = Bt + (size_t)(t * 16 + rl) * K + k + half * 8;
      V16BF b;
      b.q[0] = *(const uint4*)(brow);
      b.q[1] = *(const uint4*)(brow + 16);
      acc[t] = __builtin_amdgcn_wmma_f32_16x16x32_bf16(false, a.v, false, b.v,
                                                       (short)0, acc[t], false, false);
    }
  }

#pragma unroll
  for (int t = 0; t < MT; ++t) {
    const int col = t * 16 + rl;
    const float bc = bias ? bias[col] : 0.f;
#pragma unroll
    for (int r = 0; r < 8; ++r) {             // D layout: M = r + 8*half, Ncol = rl
      float v = acc[t][r] + bc;
      if (act) v = leaky(v);
      C[(size_t)(row0 + r + half * 8) * Mo + col] = v;
    }
  }
}

// ---------------- elementwise / graph kernels ----------------
__global__ void fill_f32(float* __restrict__ p, float v, int n) {
  int i = blockIdx.x * blockDim.x + threadIdx.x;
  if (i < n) p[i] = v;
}

// bn1d(x)*scale+shift -> bf16, zero-padded K -> Kp
__global__ void prep_bn_bf16(const float* __restrict__ X, const float* __restrict__ g,
                             const float* __restrict__ b, unsigned short* __restrict__ out,
                             int N, int K, int Kp) {
  int i = blockIdx.x * blockDim.x + threadIdx.x;
  if (i >= N * Kp) return;
  int k = i % Kp;
  float v = 0.f;
  if (k < K) {
    int row = i / Kp;
    v = X[(size_t)row * K + k] * (g[k] * BN_SCALE) + b[k];
  }
  out[i] = f2bf(v);
}

// bn1d(l2norm_row(x)) -> bf16   (one thread per row)
__global__ void prep_ln_bn_bf16(const float* __restrict__ X, const float* __restrict__ g,
                                const float* __restrict__ b, unsigned short* __restrict__ out,
                                int N, int C) {
  int row = blockIdx.x * blockDim.x + threadIdx.x;
  if (row >= N) return;
  const float* xr = X + (size_t)row * C;
  float ss = 0.f;
  for (int c = 0; c < C; ++c) { float v = xr[c]; ss += v * v; }
  float inv = 1.f / fmaxf(sqrtf(ss), 1e-12f);
  unsigned short* orow = out + (size_t)row * C;
  for (int c = 0; c < C; ++c)
    orow[c] = f2bf(xr[c] * inv * (g[c] * BN_SCALE) + b[c]);
}

// W[K,Mo] fp32 -> Wt[Mo,Kp] bf16 (transposed, zero-padded)
__global__ void prep_wt_bf16(const float* __restrict__ Wm, unsigned short* __restrict__ Wt,
                             int K, int Kp, int Mo) {
  int i = blockIdx.x * blockDim.x + threadIdx.x;
  if (i >= Mo * Kp) return;
  int m = i / Kp, k = i % Kp;
  Wt[i] = f2bf(k < K ? Wm[(size_t)k * Mo + m] : 0.f);
}

__global__ void deg_scatter(const int* __restrict__ dst, const float* __restrict__ ew,
                            float* __restrict__ deg, int nE) {
  int e = blockIdx.x * blockDim.x + threadIdx.x;
  if (e < nE) atomicAdd(&deg[dst[e]], ew[e]);
}

__global__ void rsqrt_pos(const float* __restrict__ deg, float* __restrict__ dinv, int n) {
  int i = blockIdx.x * blockDim.x + threadIdx.x;
  if (i < n) { float d = deg[i]; dinv[i] = d > 0.f ? rsqrtf(d) : 0.f; }
}

__global__ void gcn_scatter(const int* __restrict__ src, const int* __restrict__ dst,
                            const float* __restrict__ ew, const float* __restrict__ dinv,
                            const float* __restrict__ XW, float* __restrict__ out,
                            int nE, int C) {
  long long i = (long long)blockIdx.x * blockDim.x + threadIdx.x;
  if (i >= (long long)nE * C) return;
  int e = (int)(i / C), c = (int)(i % C);
  int s = src[e], d = dst[e];
  float nrm = dinv[s] * ew[e] * dinv[d];
  atomicAdd(&out[(size_t)d * C + c], XW[(size_t)s * C + c] * nrm);
}

// agg += self-loop (xw * dinv^2) + bias ; leaky ; in-place
__global__ void gcn_finalize(float* __restrict__ agg, const float* __restrict__ XW,
                             const float* __restrict__ dinv, const float* __restrict__ bias,
                             int N, int C) {
  int i = blockIdx.x * blockDim.x + threadIdx.x;
  if (i >= N * C) return;
  int row = i / C, c = i % C;
  float di = dinv[row];
  agg[i] = leaky(agg[i] + XW[i] * di * di + bias[c]);
}

__global__ void q_assign(const float* __restrict__ Q, int* __restrict__ assign, int N, int S) {
  int p = blockIdx.x * blockDim.x + threadIdx.x;
  if (p >= N) return;
  const float* qr = Q + (size_t)p * S;
  int best = 0; float bv = qr[0];
  for (int j = 1; j < S; ++j) { float q = qr[j]; if (q > bv) { bv = q; best = j; } }
  assign[p] = best;
}

__global__ void pool_accum(const float* __restrict__ X, const int* __restrict__ assign,
                           float* __restrict__ sum, float* __restrict__ cnt, int N, int C) {
  int i = blockIdx.x * blockDim.x + threadIdx.x;
  if (i >= N * C) return;
  int p = i / C, c = i % C;
  int a = assign[p];
  atomicAdd(&sum[(size_t)a * C + c], X[i]);
  if (c == 0) atomicAdd(&cnt[a], 1.f);
}

__global__ void pool_div(float* __restrict__ sum, const float* __restrict__ cnt, int S, int C) {
  int i = blockIdx.x * blockDim.x + threadIdx.x;
  if (i >= S * C) return;
  float c = cnt[i / C];
  sum[i] = c > 0.f ? sum[i] / c : 0.f;
}

// depthwise 5x5, pad 2, channels-last [H,W,C]; + bias; leaky
__global__ void dwconv5(const float* __restrict__ in, const float* __restrict__ k5,
                        const float* __restrict__ bias, float* __restrict__ out,
                        int H, int W, int C) {
  int i = blockIdx.x * blockDim.x + threadIdx.x;
  if (i >= H * W * C) return;
  int c = i % C, x = (i / C) % W, y = i / (C * W);
  const float* kc = k5 + c * 25;
  float acc = 0.f;
  for (int ky = 0; ky < 5; ++ky) {
    int yy = y + ky - 2; if (yy < 0 || yy >= H) continue;
    for (int kx = 0; kx < 5; ++kx) {
      int xx = x + kx - 2; if (xx < 0 || xx >= W) continue;
      acc += in[((size_t)yy * W + xx) * C + c] * kc[ky * 5 + kx];
    }
  }
  out[i] = leaky(acc + bias[c]);
}

__global__ void sgc_rowdeg(const float* __restrict__ A, float* __restrict__ dinv, int S) {
  int i = blockIdx.x * blockDim.x + threadIdx.x;
  if (i >= S) return;
  const float* r = A + (size_t)i * S;
  float s = 1.f;                       // + self loop
  for (int j = 0; j < S; ++j) s += r[j];
  dinv[i] = rsqrtf(s);
}

__global__ void sgc_ahat(const float* __restrict__ A, const float* __restrict__ dinv,
                         unsigned short* __restrict__ out, int S) {
  int idx = blockIdx.x * blockDim.x + threadIdx.x;
  if (idx >= S * S) return;
  int i = idx / S, j = idx % S;
  float v = A[idx] + (i == j ? 1.f : 0.f);
  out[idx] = f2bf(v * dinv[i] * dinv[j]);
}

// Z[N,M] fp32 -> Zt[M,N] bf16
__global__ void transpose_bf16(const float* __restrict__ Z, unsigned short* __restrict__ Zt,
                               int N, int M) {
  int i = blockIdx.x * blockDim.x + threadIdx.x;
  if (i >= N * M) return;
  int n = i / M, m = i % M;
  Zt[(size_t)m * N + n] = f2bf(Z[i]);
}

// concat(Hs[assign]+pg, lse) -> bf16 [N,128]
__global__ void final_prep(const float* __restrict__ Hs, const int* __restrict__ assign,
                           const float* __restrict__ pg, const float* __restrict__ lse,
                           unsigned short* __restrict__ out, int N) {
  int i = blockIdx.x * blockDim.x + threadIdx.x;
  if (i >= N * 128) return;
  int p = i >> 7, c = i & 127;
  float v;
  if (c < 64) v = Hs[(size_t)assign[p] * 64 + c] + pg[(size_t)p * 64 + c];
  else        v = lse[(size_t)p * 64 + (c - 64)];
  out[i] = f2bf(v);
}

__global__ void softmax16(const float* __restrict__ Y, float* __restrict__ out, int N) {
  int p = blockIdx.x * blockDim.x + threadIdx.x;
  if (p >= N) return;
  const float* yr = Y + (size_t)p * 16;
  float m = yr[0];
  for (int j = 1; j < 16; ++j) m = fmaxf(m, yr[j]);
  float e[16], s = 0.f;
  for (int j = 0; j < 16; ++j) { e[j] = expf(yr[j] - m); s += e[j]; }
  float inv = 1.f / s;
  for (int j = 0; j < 16; ++j) out[(size_t)p * 16 + j] = e[j] * inv;
}

// ---------------- host orchestration ----------------
static inline unsigned cdiv(long long n, int b) { return (unsigned)((n + b - 1) / b); }

extern "C" void kernel_launch(void* const* d_in, const int* in_sizes, int n_in,
                              void* d_out, int out_size, void* d_ws, size_t ws_size,
                              hipStream_t stream) {
  (void)in_sizes; (void)n_in; (void)out_size; (void)ws_size;
  const int Himg = 256, Wimg = 256, CIN = 144;
  const int N = Himg * Wimg, S = 1024, NE = 262144;
  const int K0P = 160;                 // 144 padded to multiple of 32

  // ---- inputs (setup_inputs dict order, params flattened in insertion order)
  const float* x      = (const float*)d_in[0];
  const float* Q      = (const float*)d_in[1];
  const float* Aadj   = (const float*)d_in[2];
  const int*   ei     = (const int*)  d_in[3];
  const float* ew     = (const float*)d_in[4];
  const float* bn0_g  = (const float*)d_in[5];
  const float* bn0_b  = (const float*)d_in[6];
  const float* W0     = (const float*)d_in[7];
  const float* b0     = (const float*)d_in[8];
  const float* bn1_g  = (const float*)d_in[9];
  const float* bn1_b  = (const float*)d_in[10];
  const float* W1     = (const float*)d_in[11];
  const float* b1     = (const float*)d_in[12];
  const float* bnL1_g = (const float*)d_in[13];
  const float* bnL1_b = (const float*)d_in[14];
  const float* pw1    = (const float*)d_in[15];
  const float* dw1    = (const float*)d_in[16];
  const float* dwb1   = (const float*)d_in[17];
  const float* bnL2_g = (const float*)d_in[18];
  const float* bnL2_b = (const float*)d_in[19];
  const float* pw2    = (const float*)d_in[20];
  const float* dw2    = (const float*)d_in[21];
  const float* dwb2   = (const float*)d_in[22];
  const float* bnS1_g = (const float*)d_in[23];
  const float* bnS1_b = (const float*)d_in[24];
  const float* Ws1    = (const float*)d_in[25];
  const float* bs1    = (const float*)d_in[26];
  const float* bnS2_g = (const float*)d_in[27];
  const float* bnS2_b = (const float*)d_in[28];
  const float* Ws2    = (const float*)d_in[29];
  const float* bs2    = (const float*)d_in[30];
  const float* bnP1_g = (const float*)d_in[31];
  const float* bnP1_b = (const float*)d_in[32];
  const float* Wp1    = (const float*)d_in[33];
  const float* bp1    = (const float*)d_in[34];
  const float* bnP2_g = (const float*)d_in[35];
  const float* bnP2_b = (const float*)d_in[36];
  const float* Wp2    = (const float*)d_in[37];
  const float* bp2    = (const float*)d_in[38];
  const float* Wsm    = (const float*)d_in[39];
  const float* bsm    = (const float*)d_in[40];

  const int* src = ei;
  const int* dst = ei + NE;

  // ---- workspace bump allocator (256B aligned)
  char* base = (char*)d_ws;
  size_t off = 0;
  auto alloc = [&](size_t bytes) -> char* {
    char* p = base + off;
    off += (bytes + 255) & ~(size_t)255;
    return p;
  };
  unsigned short* Abuf = (unsigned short*)alloc((size_t)N * K0P * 2);  // GEMM A operand
  unsigned short* Wbuf = (unsigned short*)alloc((size_t)128 * K0P * 2);
  unsigned short* Ahat = (unsigned short*)alloc((size_t)S * S * 2);
  unsigned short* Ztb  = (unsigned short*)alloc((size_t)128 * S * 2);
  unsigned short* SAbf = (unsigned short*)alloc((size_t)S * 128 * 2);
  float* F0 = (float*)alloc((size_t)N * 128 * 4);   // t / z1 / z2
  float* F1 = (float*)alloc((size_t)N * 128 * 4);   // clean
  float* F2 = (float*)alloc((size_t)N * 128 * 4);   // l1
  float* F3 = (float*)alloc((size_t)N * 128 * 4);   // xw / xw2
  float* F4 = (float*)alloc((size_t)N * 128 * 4);   // pg1
  float* F5 = (float*)alloc((size_t)N * 64 * 4);    // pg
  float* F6 = (float*)alloc((size_t)N * 64 * 4);    // lse_res
  float* deg   = (float*)alloc((size_t)N * 4);
  float* dinv  = (float*)alloc((size_t)N * 4);
  int*   assn  = (int*)  alloc((size_t)N * 4);
  float* cnt   = (float*)alloc((size_t)S * 4);
  float* H0    = (float*)alloc((size_t)S * 128 * 4);
  float* SZ    = (float*)alloc((size_t)S * 128 * 4);
  float* SH1   = (float*)alloc((size_t)S * 128 * 4);
  float* SZ2   = (float*)alloc((size_t)S * 64 * 4);
  float* SH2   = (float*)alloc((size_t)S * 64 * 4);
  float* dinvS = (float*)alloc((size_t)S * 4);
  float* Ybuf  = (float*)alloc((size_t)N * 16 * 4);

  const dim3 B256(256);

  // ---- Stage 1: t = leaky(bn0(x) @ W0 + b0)
  prep_bn_bf16<<<cdiv((long long)N * K0P, 256), B256, 0, stream>>>(x, bn0_g, bn0_b, Abuf, N, CIN, K0P);
  prep_wt_bf16<<<cdiv(128 * K0P, 256), B256, 0, stream>>>(W0, Wbuf, CIN, K0P, 128);
  wmma_gemm_mt<8><<<N / 128, B256, 0, stream>>>(Abuf, Wbuf, b0, F0, N, K0P, 1);

  // ---- Stage 2: clean = leaky(bn1(t) @ W1 + b1)
  prep_bn_bf16<<<cdiv((long long)N * 128, 256), B256, 0, stream>>>(F0, bn1_g, bn1_b, Abuf, N, 128, 128);
  prep_wt_bf16<<<cdiv(128 * 128, 256), B256, 0, stream>>>(W1, Wbuf, 128, 128, 128);
  wmma_gemm_mt<8><<<N / 128, B256, 0, stream>>>(Abuf, Wbuf, b1, F1, N, 128, 1);

  // ---- GCN degree (shared by both PGC layers; self-loop weight 1)
  fill_f32<<<cdiv(N, 256), B256, 0, stream>>>(deg, 1.f, N);
  deg_scatter<<<cdiv(NE, 256), B256, 0, stream>>>(dst, ew, deg, NE);
  rsqrt_pos<<<cdiv(N, 256), B256, 0, stream>>>(deg, dinv, N);

  // ---- PGC layer 1 (128 -> 128)
  prep_ln_bn_bf16<<<cdiv(N, 256), B256, 0, stream>>>(F1, bnP1_g, bnP1_b, Abuf, N, 128);
  prep_wt_bf16<<<cdiv(128 * 128, 256), B256, 0, stream>>>(Wp1, Wbuf, 128, 128, 128);
  wmma_gemm_mt<8><<<N / 128, B256, 0, stream>>>(Abuf, Wbuf, nullptr, F3, N, 128, 0);
  fill_f32<<<cdiv((long long)N * 128, 256), B256, 0, stream>>>(F4, 0.f, N * 128);
  gcn_scatter<<<cdiv((long long)NE * 128, 256), B256, 0, stream>>>(src, dst, ew, dinv, F3, F4, NE, 128);
  gcn_finalize<<<cdiv((long long)N * 128, 256), B256, 0, stream>>>(F4, F3, dinv, bp1, N, 128);

  // ---- PGC layer 2 (128 -> 64)
  prep_ln_bn_bf16<<<cdiv(N, 256), B256, 0, stream>>>(F4, bnP2_g, bnP2_b, Abuf, N, 128);
  prep_wt_bf16<<<cdiv(64 * 128, 256), B256, 0, stream>>>(Wp2, Wbuf, 128, 128, 64);
  wmma_gemm_mt<4><<<N / 128, B256, 0, stream>>>(Abuf, Wbuf, nullptr, F3, N, 128, 0);
  fill_f32<<<cdiv((long long)N * 64, 256), B256, 0, stream>>>(F5, 0.f, N * 64);
  gcn_scatter<<<cdiv((long long)NE * 64, 256), B256, 0, stream>>>(src, dst, ew, dinv, F3, F5, NE, 64);
  gcn_finalize<<<cdiv((long long)N * 64, 256), B256, 0, stream>>>(F5, F3, dinv, bp2, N, 64);

  // ---- superpixel mean pooling of clean
  q_assign<<<cdiv(N, 256), B256, 0, stream>>>(Q, assn, N, S);
  fill_f32<<<cdiv(S * 128, 256), B256, 0, stream>>>(H0, 0.f, S * 128);
  fill_f32<<<cdiv(S, 256), B256, 0, stream>>>(cnt, 0.f, S);
  pool_accum<<<cdiv((long long)N * 128, 256), B256, 0, stream>>>(F1, assn, H0, cnt, N, 128);
  pool_div<<<cdiv(S * 128, 256), B256, 0, stream>>>(H0, cnt, S, 128);

  // ---- LSE block 1: bnL1 -> pw1 (leaky) -> dw 5x5 (+dwb1, leaky)
  prep_bn_bf16<<<cdiv((long long)N * 128, 256), B256, 0, stream>>>(F1, bnL1_g, bnL1_b, Abuf, N, 128, 128);
  prep_wt_bf16<<<cdiv(128 * 128, 256), B256, 0, stream>>>(pw1, Wbuf, 128, 128, 128);
  wmma_gemm_mt<8><<<N / 128, B256, 0, stream>>>(Abuf, Wbuf, nullptr, F0, N, 128, 1);
  dwconv5<<<cdiv((long long)N * 128, 256), B256, 0, stream>>>(F0, dw1, dwb1, F2, Himg, Wimg, 128);

  // ---- LSE block 2: bnL2 -> pw2 (leaky) -> dw 5x5 (+dwb2, leaky)
  prep_bn_bf16<<<cdiv((long long)N * 128, 256), B256, 0, stream>>>(F2, bnL2_g, bnL2_b, Abuf, N, 128, 128);
  prep_wt_bf16<<<cdiv(64 * 128, 256), B256, 0, stream>>>(pw2, Wbuf, 128, 128, 64);
  wmma_gemm_mt<4><<<N / 128, B256, 0, stream>>>(Abuf, Wbuf, nullptr, F0, N, 128, 1);
  dwconv5<<<cdiv((long long)N * 64, 256), B256, 0, stream>>>(F0, dw2, dwb2, F6, Himg, Wimg, 64);

  // ---- SGC on normalized superpixel adjacency
  sgc_rowdeg<<<cdiv(S, 256), B256, 0, stream>>>(Aadj, dinvS, S);
  sgc_ahat<<<cdiv(S * S, 256), B256, 0, stream>>>(Aadj, dinvS, Ahat, S);
  // layer 1: Hs1 = leaky(Ahat @ (bnS1(l2norm(H0)) @ Ws1 + bs1))
  prep_ln_bn_bf16<<<cdiv(S, 256), B256, 0, stream>>>(H0, bnS1_g, bnS1_b, SAbf, S, 128);
  prep_wt_bf16<<<cdiv(128 * 128, 256), B256, 0, stream>>>(Ws1, Wbuf, 128, 128, 128);
  wmma_gemm_mt<8><<<S / 128, B256, 0, stream>>>(SAbf, Wbuf, bs1, SZ, S, 128, 0);
  transpose_bf16<<<cdiv(S * 128, 256), B256, 0, stream>>>(SZ, Ztb, S, 128);
  wmma_gemm_mt<8><<<S / 128, B256, 0, stream>>>(Ahat, Ztb, nullptr, SH1, S, S, 1);
  // layer 2: Hs2 = leaky(Ahat @ (bnS2(l2norm(Hs1)) @ Ws2 + bs2))
  prep_ln_bn_bf16<<<cdiv(S, 256), B256, 0, stream>>>(SH1, bnS2_g, bnS2_b, SAbf, S, 128);
  prep_wt_bf16<<<cdiv(64 * 128, 256), B256, 0, stream>>>(Ws2, Wbuf, 128, 128, 64);
  wmma_gemm_mt<4><<<S / 128, B256, 0, stream>>>(SAbf, Wbuf, bs2, SZ2, S, 128, 0);
  transpose_bf16<<<cdiv(S * 64, 256), B256, 0, stream>>>(SZ2, Ztb, S, 64);
  wmma_gemm_mt<4><<<S / 128, B256, 0, stream>>>(Ahat, Ztb, nullptr, SH2, S, S, 1);

  // ---- decode + classifier + softmax
  final_prep<<<cdiv((long long)N * 128, 256), B256, 0, stream>>>(SH2, assn, F5, F6, Abuf, N);
  prep_wt_bf16<<<cdiv(16 * 128, 256), B256, 0, stream>>>(Wsm, Wbuf, 128, 128, 16);
  wmma_gemm_mt<1><<<N / 128, B256, 0, stream>>>(Abuf, Wbuf, bsm, Ybuf, N, 128, 0);
  softmax16<<<cdiv(N, 256), B256, 0, stream>>>(Ybuf, (float*)d_out, N);
}
